// DeformBlock_27453430956037
// MI455X (gfx1250) — matrix operand order repaired and checked by reading, not compiled, not run on hardware
//
#include <hip/hip_runtime.h>
#include <hip/hip_bf16.h>

// ---------------------------------------------------------------------------
// Deformable conv block for gfx1250 (MI455X, wave32, WMMA).
// Pipeline: transpose x -> NHWC, offset/mask 3x3 conv, bf16 weight repack,
// fused bilinear-sample + bf16 WMMA GEMM (fp32 accumulate).
// ---------------------------------------------------------------------------

typedef __bf16 bf16x16 __attribute__((ext_vector_type(16)));
typedef float floatx8 __attribute__((ext_vector_type(8)));

union FragU {
    uint4 u[2];
    bf16x16 v;
};

#define CH 128
#define HH 64
#define WW 64
#define HW 4096
#define KK 9
#define KDIM 1152  // CH * KK

__device__ __forceinline__ unsigned int pack2bf(float a, float b) {
    unsigned int ua = __float_as_uint(a);
    ua += 0x7FFFu + ((ua >> 16) & 1u);          // round-to-nearest-even
    unsigned int ub = __float_as_uint(b);
    ub += 0x7FFFu + ((ub >> 16) & 1u);
    return (ua >> 16) | (ub & 0xFFFF0000u);
}

__device__ __forceinline__ float4 fma4(float4 a, float s, float4 acc) {
    acc.x += a.x * s; acc.y += a.y * s; acc.z += a.z * s; acc.w += a.w * s;
    return acc;
}

// -------------------- NCHW -> NHWC transpose (per batch) -------------------
__global__ void transpose_nchw_nhwc(const float* __restrict__ x,
                                    float* __restrict__ xT) {
    __shared__ float tile[32][33];
    const int b  = blockIdx.z;
    const int c0 = blockIdx.y * 32;
    const int p0 = blockIdx.x * 32;
    const int tx = threadIdx.x, ty = threadIdx.y;   // 32 x 8
    const float* xb = x + (size_t)b * CH * HW;
#pragma unroll
    for (int i = 0; i < 32; i += 8)
        tile[ty + i][tx] = xb[(size_t)(c0 + ty + i) * HW + p0 + tx];
    __syncthreads();
    float* xo = xT + (size_t)b * HW * CH;
#pragma unroll
    for (int i = 0; i < 32; i += 8)
        xo[(size_t)(p0 + ty + i) * CH + c0 + tx] = tile[tx][ty + i];
}

// -------------------- weight (C,C,3,3) fp32 -> bf16 [O][kk*128+c] ----------
__global__ void weight_prep(const float* __restrict__ wgt,
                            unsigned short* __restrict__ Wb) {
    int i = blockIdx.x * 256 + threadIdx.x;
    if (i >= CH * KDIM) return;
    int o  = i / KDIM;
    int r  = i - o * KDIM;
    int kk = r >> 7;
    int c  = r & 127;
    float v = wgt[((size_t)o * CH + c) * KK + kk];
    unsigned int u = __float_as_uint(v);
    u += 0x7FFFu + ((u >> 16) & 1u);
    Wb[i] = (unsigned short)(u >> 16);
}

// -------------------- offset + mask 3x3 conv (27 outputs/pixel) ------------
__global__ void offmask_conv(const float* __restrict__ xT,
                             const float* __restrict__ offw,
                             const float* __restrict__ offb,
                             const float* __restrict__ mw,
                             const float* __restrict__ mb,
                             float* __restrict__ offs,
                             float* __restrict__ maskb) {
    int idx = blockIdx.x * blockDim.x + threadIdx.x;   // 0..32767
    int b  = idx >> 12;
    int hw = idx & (HW - 1);
    int h  = hw >> 6, w = hw & 63;

    float acc[27];
#pragma unroll
    for (int co = 0; co < 18; ++co) acc[co] = offb[co];
#pragma unroll
    for (int co = 0; co < 9; ++co)  acc[18 + co] = mb[co];

    const float* xb = xT + (size_t)b * HW * CH;
#pragma unroll 1
    for (int tap = 0; tap < 9; ++tap) {
        int y = h + tap / 3 - 1;
        int x = w + tap % 3 - 1;
        if (y < 0 || y >= HH || x < 0 || x >= WW) continue;
        const float* xv = xb + ((y << 6) + x) * CH;
        const float* wo = offw + tap;   // + c*9 + co*1152
        const float* wm = mw + tap;
#pragma unroll 1
        for (int c = 0; c < CH; ++c) {
            float v = xv[c];
#pragma unroll
            for (int co = 0; co < 18; ++co)
                acc[co] += v * wo[(size_t)co * KDIM + (size_t)c * KK];
#pragma unroll
            for (int co = 0; co < 9; ++co)
                acc[18 + co] += v * wm[(size_t)co * KDIM + (size_t)c * KK];
        }
    }
#pragma unroll
    for (int co = 0; co < 18; ++co)
        offs[(((size_t)b * 18 + co) << 12) + hw] = acc[co];
#pragma unroll
    for (int co = 0; co < 9; ++co)
        maskb[(((size_t)b * 9 + co) << 12) + hw] =
            1.0f / (1.0f + __expf(-acc[18 + co]));
}

// -------------------- fused bilinear sample + WMMA GEMM --------------------
// grid (64 rows, 8 batches), block 256 = 8 waves. Per block: out tile
// [128 O][64 pixels] for one image row of one batch.
__global__ __launch_bounds__(256, 1)
void deform_gemm(const float* __restrict__ xT,
                 const float* __restrict__ offs,
                 const float* __restrict__ maskb,
                 const unsigned short* __restrict__ Wb,
                 const float* __restrict__ bias,
                 float* __restrict__ out) {
    __shared__ int   sA[64][4];
    __shared__ float sW[64][4];
    __shared__ __attribute__((aligned(16))) unsigned int S[64][68]; // 272B rows

    const int t    = threadIdx.x;
    const int lane = t & 31;
    const int wid  = t >> 5;
    const int Mbase = wid << 4;
    const int h = blockIdx.x;
    const int b = blockIdx.y;
    const float* xb = xT + ((size_t)b << 12) * CH;

    floatx8 acc[4] = {};   // 4 N-subtiles of 16x16 fp32

#pragma unroll 1
    for (int kk = 0; kk < KK; ++kk) {
        // --- per-pixel sampling setup: addresses + bilinear*mask weights ---
        if (t < 64) {
            const int w  = t;
            const int hw = (h << 6) + w;
            float dy = offs[(((size_t)b * 18 + 2 * kk) << 12) + hw];
            float dx = offs[(((size_t)b * 18 + 2 * kk + 1) << 12) + hw];
            float m  = maskb[(((size_t)b * 9 + kk) << 12) + hw];
            float py = (float)(h + kk / 3 - 1) + dy;
            float px = (float)(w + kk % 3 - 1) + dx;
            float y0f = floorf(py), x0f = floorf(px);
            int   y0 = (int)y0f,   x0 = (int)x0f;
            float ly = py - y0f,   lx = px - x0f;
            float wy[2] = {1.0f - ly, ly};
            float wx[2] = {1.0f - lx, lx};
#pragma unroll
            for (int j = 0; j < 4; ++j) {
                int yi = y0 + (j >> 1), xi = x0 + (j & 1);
                bool ok = (yi >= 0) && (yi < HH) && (xi >= 0) && (xi < WW);
                int yc = min(max(yi, 0), HH - 1);
                int xc = min(max(xi, 0), WW - 1);
                sA[w][j] = (((yc << 6) + xc) << 7);   // element offset in xT slice
                sW[w][j] = ok ? wy[j >> 1] * wx[j & 1] * m : 0.0f;
            }
        }
        __syncthreads();   // prev GEMM done + sampling data visible

        // --- build S[pixel][channel] bf16 tile: 32 ch per thread ---
        {
            const int p  = t & 63;
            const int cg = t >> 6;           // 0..3 -> 32-channel group
            const int a0 = sA[p][0], a1 = sA[p][1], a2 = sA[p][2], a3 = sA[p][3];
            const float w0 = sW[p][0], w1 = sW[p][1], w2 = sW[p][2], w3 = sW[p][3];
#pragma unroll
            for (int cc = 0; cc < 32; cc += 8) {
                const int c = (cg << 5) + cc;
                float4 r0 = {0, 0, 0, 0}, r1 = {0, 0, 0, 0};
                r0 = fma4(*(const float4*)(xb + a0 + c), w0, r0);
                r1 = fma4(*(const float4*)(xb + a0 + c + 4), w0, r1);
                r0 = fma4(*(const float4*)(xb + a1 + c), w1, r0);
                r1 = fma4(*(const float4*)(xb + a1 + c + 4), w1, r1);
                r0 = fma4(*(const float4*)(xb + a2 + c), w2, r0);
                r1 = fma4(*(const float4*)(xb + a2 + c + 4), w2, r1);
                r0 = fma4(*(const float4*)(xb + a3 + c), w3, r0);
                r1 = fma4(*(const float4*)(xb + a3 + c + 4), w3, r1);
                uint4 pk;
                pk.x = pack2bf(r0.x, r0.y);
                pk.y = pack2bf(r0.z, r0.w);
                pk.z = pack2bf(r1.x, r1.y);
                pk.w = pack2bf(r1.z, r1.w);
                *(uint4*)&S[p][(cg << 4) + (cc >> 1)] = pk;
            }
        }
        __syncthreads();   // S tile visible

        // --- WMMA: this wave's 16 O-rows x 64 pixels, K-chunk = this tap ---
#pragma unroll
        for (int cb = 0; cb < 4; ++cb) {
            const int kb   = kk * CH + (cb << 5);
            const int row  = Mbase + (lane & 15);
            const int koff = kb + ((lane >> 4) << 3);
            // A fragment: lane<16 -> K {0..7,16..23}; lane>=16 -> {8..15,24..31}
            const uint4* wrow =
                (const uint4*)(Wb + (size_t)row * KDIM + koff);
            FragU afr;
            afr.u[0] = wrow[0];   // 8 bf16: K koff..koff+7
            afr.u[1] = wrow[2];   // 8 bf16: K koff+16..koff+23
#pragma unroll
            for (int n0 = 0; n0 < 4; ++n0) {
                const int p  = (n0 << 4) + (lane & 15);
                const int cu = (cb << 4) + ((lane >> 4) << 2); // uint col
                FragU bfr;
                bfr.u[0] = *(const uint4*)&S[p][cu];
                bfr.u[1] = *(const uint4*)&S[p][cu + 8];
                acc[n0] = __builtin_amdgcn_wmma_f32_16x16x32_bf16(
                    false, afr.v, false, bfr.v, (short)0, acc[n0],
                    false, false);
            }
        }
    }

    // --- epilogue: D layout lanes 0-15 -> M=r, lanes 16-31 -> M=r+8 ---
    const int obase = Mbase + ((lane >> 4) << 3);
#pragma unroll
    for (int n0 = 0; n0 < 4; ++n0) {
        const int pix = (h << 6) + (n0 << 4) + (lane & 15);
#pragma unroll
        for (int r = 0; r < 8; ++r) {
            const int o = obase + r;
            out[(((size_t)b * CH + o) << 12) + pix] = acc[n0][r] + bias[o];
        }
    }
}

// ---------------------------------------------------------------------------
extern "C" void kernel_launch(void* const* d_in, const int* in_sizes, int n_in,
                              void* d_out, int out_size, void* d_ws, size_t ws_size,
                              hipStream_t stream) {
    const float* x        = (const float*)d_in[0];   // (8,128,64,64)
    const float* offset_w = (const float*)d_in[1];   // (18,128,3,3)
    const float* offset_b = (const float*)d_in[2];   // (18,)
    const float* mask_w   = (const float*)d_in[3];   // (9,128,3,3)
    const float* mask_b   = (const float*)d_in[4];   // (9,)
    const float* weight   = (const float*)d_in[5];   // (128,128,3,3)
    const float* bias     = (const float*)d_in[6];   // (128,)
    float* out = (float*)d_out;                      // (8,128,64,64)

    char* ws = (char*)d_ws;
    float* xT            = (float*)(ws);                       // 16,777,216 B
    float* offs          = (float*)(ws + 16777216);            //  2,359,296 B
    float* maskb         = (float*)(ws + 16777216 + 2359296);  //  1,179,648 B
    unsigned short* Wb   = (unsigned short*)(ws + 16777216 + 2359296 + 1179648);
                                                               //    294,912 B

    transpose_nchw_nhwc<<<dim3(128, 4, 8), dim3(32, 8), 0, stream>>>(x, xT);
    weight_prep<<<(CH * KDIM + 255) / 256, 256, 0, stream>>>(weight, Wb);
    offmask_conv<<<128, 256, 0, stream>>>(xT, offset_w, offset_b,
                                          mask_w, mask_b, offs, maskb);
    deform_gemm<<<dim3(64, 8), 256, 0, stream>>>(xT, offs, maskb, Wb, bias, out);
}
